// CrossModalFusion_618475290713
// MI455X (gfx1250) — compile-verified, matrix operand order checked by reading, not run on hardware
//
#include <hip/hip_runtime.h>
#include <hip/hip_bf16.h>
#include <stdint.h>
#include <stddef.h>

// ---------------------------------------------------------------------------
// CDNA5 (gfx1250) fused cross-modal attention, bf16 WMMA pipeline, round 3:
//  - attention K/V tiles staged into LDS once per block via
//    GLOBAL_LOAD_ASYNC_TO_LDS_B128 (ASYNCcnt-tracked, double buffered),
//    removing the 4x redundant per-wave global loads
//  - GEMMs unchanged: 32x64 wave tile, 8 WMMAs per 32-deep K step, all
//    operands 16B vector loads (weights pre-transposed, V stored transposed)
// ---------------------------------------------------------------------------

typedef __attribute__((ext_vector_type(16))) __bf16 bf16x16;
typedef __attribute__((ext_vector_type(8)))  __bf16 bf16x8;
typedef __attribute__((ext_vector_type(8)))  float  f32x8;

union FragU { bf16x16 v; bf16x8 h[2]; };

__device__ __forceinline__ f32x8 wmma_bf16(bf16x16 a, bf16x16 b, f32x8 c) {
  // (neg_a, A, neg_b, B, c_mod, C, reuse_a, reuse_b)
  return __builtin_amdgcn_wmma_f32_16x16x32_bf16(false, a, false, b, (short)0, c,
                                                 false, false);
}

// LDS byte offset (addrspace(3) pointer value) of a __shared__ object.
__device__ __forceinline__ unsigned lds_off(const void* p) {
  return (unsigned)(size_t)(__attribute__((address_space(3))) const char*)p;
}

// Async DMA: 16 bytes global -> LDS per enabled lane; tracked by ASYNCcnt.
__device__ __forceinline__ void async_ld_b128(unsigned lds_byte_off, const void* g) {
  asm volatile("global_load_async_to_lds_b128 %0, %1, off"
               :: "v"(lds_byte_off), "v"(g) : "memory");
}
__device__ __forceinline__ void wait_async0() {
  asm volatile("s_wait_asynccnt 0" ::: "memory");
}

#define DINO  768
#define COG   1920
#define NHEAD 12
#define HD    64
#define BHW   1024   // H*W = 32*32
#define TKV   4096   // T*H*W = 4*32*32
#define NB    2

// ---------------------------------------------------------------------------
// LayerNorm for dino_query [B, C=768, H, W] (channel-first, stride-1024 gather)
// -> bf16 rows [B*HW, 768]
// ---------------------------------------------------------------------------
__global__ __launch_bounds__(256) void ln_dino_kernel(
    const float* __restrict__ x, const float* __restrict__ g,
    const float* __restrict__ be, __bf16* __restrict__ y) {
  const int row = blockIdx.x;               // [0, NB*BHW)
  const int b  = row >> 10;
  const int hw = row & 1023;
  const float* base = x + (size_t)b * DINO * BHW + hw;
  __shared__ float r1[256], r2[256];
  float s = 0.f, s2 = 0.f;
  for (int c = threadIdx.x; c < DINO; c += 256) {
    float v = base[(size_t)c * BHW];
    s += v; s2 += v * v;
  }
  r1[threadIdx.x] = s; r2[threadIdx.x] = s2;
  __syncthreads();
  for (int off = 128; off > 0; off >>= 1) {
    if ((int)threadIdx.x < off) {
      r1[threadIdx.x] += r1[threadIdx.x + off];
      r2[threadIdx.x] += r2[threadIdx.x + off];
    }
    __syncthreads();
  }
  const float mean = r1[0] * (1.f / DINO);
  const float var  = r2[0] * (1.f / DINO) - mean * mean;
  const float inv  = rsqrtf(var + 1e-5f);
  __bf16* yr = y + (size_t)row * DINO;
  for (int c = threadIdx.x; c < DINO; c += 256) {
    float v = (base[(size_t)c * BHW] - mean) * inv * g[c] + be[c];
    yr[c] = (__bf16)v;
  }
}

// ---------------------------------------------------------------------------
// LayerNorm for cogvideo_keys [B,T,H,W,C=1920] (contiguous rows)
// -> bf16 rows [B*TKV, 1920]
// ---------------------------------------------------------------------------
__global__ __launch_bounds__(256) void ln_cog_kernel(
    const float* __restrict__ x, const float* __restrict__ g,
    const float* __restrict__ be, __bf16* __restrict__ y) {
  const int row = blockIdx.x;               // [0, NB*TKV)
  const float* base = x + (size_t)row * COG;
  __shared__ float r1[256], r2[256];
  float s = 0.f, s2 = 0.f;
  for (int c = threadIdx.x; c < COG; c += 256) {
    float v = base[c];
    s += v; s2 += v * v;
  }
  r1[threadIdx.x] = s; r2[threadIdx.x] = s2;
  __syncthreads();
  for (int off = 128; off > 0; off >>= 1) {
    if ((int)threadIdx.x < off) {
      r1[threadIdx.x] += r1[threadIdx.x + off];
      r2[threadIdx.x] += r2[threadIdx.x + off];
    }
    __syncthreads();
  }
  const float mean = r1[0] * (1.f / COG);
  const float var  = r2[0] * (1.f / COG) - mean * mean;
  const float inv  = rsqrtf(var + 1e-5f);
  __bf16* yr = y + (size_t)row * COG;
  for (int c = threadIdx.x; c < COG; c += 256) {
    float v = (base[c] - mean) * inv * g[c] + be[c];
    yr[c] = (__bf16)v;
  }
}

// ---------------------------------------------------------------------------
// Transposing fp32 -> bf16 convert: in W[K,N] -> out WT[N,K].
// ---------------------------------------------------------------------------
__global__ __launch_bounds__(256) void cvtT_kernel(const float* __restrict__ in,
                                                   __bf16* __restrict__ out,
                                                   int K, int N) {
  __shared__ float t[32][33];
  const int k0 = blockIdx.x * 32;
  const int n0 = blockIdx.y * 32;
  for (int r = threadIdx.y; r < 32; r += 8)
    t[r][threadIdx.x] = in[(size_t)(k0 + r) * N + n0 + threadIdx.x];
  __syncthreads();
  for (int r = threadIdx.y; r < 32; r += 8)
    out[(size_t)(n0 + r) * K + k0 + threadIdx.x] = (__bf16)t[threadIdx.x][r];
}

// ---------------------------------------------------------------------------
// bf16 GEMM with transposed weights:
//   C = A[M,K](bf16) @ WT[N,K]^T (bf16) + bias[N](f32)
// STORE_MODE: 0 = bf16 row-major [M,N]
//             1 = bf16 V-transposed [(m>>12)*N + c][4096]  (for attention V)
//             2 = f32  [B,C,H,W] transposed final output
// ---------------------------------------------------------------------------
template <int STORE_MODE>
__global__ __launch_bounds__(128) void gemm_kernel(
    const __bf16* __restrict__ A, const __bf16* __restrict__ WT,
    const float* __restrict__ bias, void* __restrict__ Cout,
    int M, int N, int K) {
  const int wave = threadIdx.x >> 5;
  const int lane = threadIdx.x & 31;
  const int grp  = lane >> 4;          // lane half: K-group selector
  const int lr   = lane & 15;          // row (A) / col (B,C) within tile
  const int sel  = grp << 3;
  const int row0 = blockIdx.x * 128 + wave * 32;
  const int col0 = blockIdx.y * 64;

  f32x8 acc[2][4] = {};
  const __bf16* ar0 = A + (size_t)(row0 + lr) * K;
  const __bf16* ar1 = A + (size_t)(row0 + 16 + lr) * K;

  for (int k0 = 0; k0 < K; k0 += 32) {
    __builtin_prefetch(ar0 + k0 + 256, 0, 1);    // global_prefetch_b8
    __builtin_prefetch(ar1 + k0 + 256, 0, 1);
    FragU a0, a1;
    a0.h[0] = *(const bf16x8*)(ar0 + k0 + sel);
    a0.h[1] = *(const bf16x8*)(ar0 + k0 + 16 + sel);
    a1.h[0] = *(const bf16x8*)(ar1 + k0 + sel);
    a1.h[1] = *(const bf16x8*)(ar1 + k0 + 16 + sel);
#pragma unroll
    for (int nt = 0; nt < 4; ++nt) {
      const __bf16* wr = WT + (size_t)(col0 + nt * 16 + lr) * K + k0;
      FragU bfr;
      bfr.h[0] = *(const bf16x8*)(wr + sel);
      bfr.h[1] = *(const bf16x8*)(wr + 16 + sel);
      acc[0][nt] = wmma_bf16(a0.v, bfr.v, acc[0][nt]);
      acc[1][nt] = wmma_bf16(a1.v, bfr.v, acc[1][nt]);
    }
  }

#pragma unroll
  for (int rg = 0; rg < 2; ++rg) {
#pragma unroll
    for (int nt = 0; nt < 4; ++nt) {
      const int c = col0 + nt * 16 + lr;
      const float bc = bias[c];
#pragma unroll
      for (int j = 0; j < 8; ++j) {
        const int m = row0 + rg * 16 + (grp << 3) + j;   // C layout: vgpr j -> row
        const float v = acc[rg][nt][j] + bc;
        if (STORE_MODE == 0) {
          ((__bf16*)Cout)[(size_t)m * N + c] = (__bf16)v;
        } else if (STORE_MODE == 1) {
          const int bt  = m >> 12;                        // m = b*4096 + key
          const int key = m & 4095;
          ((__bf16*)Cout)[((size_t)bt * N + c) * TKV + key] = (__bf16)v;
        } else {
          const int b  = m >> 10;                         // m = b*1024 + hw
          const int hw = m & 1023;
          ((float*)Cout)[((size_t)b * N + c) * BHW + hw] = v;
        }
      }
    }
  }
}

// ---------------------------------------------------------------------------
// Flash attention with async-LDS staging.
// One block per (b, head, 64-query tile); 4 waves, each owning 16 query rows.
// All waves share (b, head), so each 16-key chunk's K tile (16x64 bf16, 2KB)
// and V^T tile (64x16 bf16, 2KB) are DMA'd into LDS exactly once per block via
// global_load_async_to_lds_b128 (one 16B move per thread per tile), double
// buffered so the next chunk's DMA overlaps this chunk's softmax + WMMAs.
// ---------------------------------------------------------------------------
__global__ __launch_bounds__(128) void attn_kernel(
    const __bf16* __restrict__ Q, const __bf16* __restrict__ Kb,
    const __bf16* __restrict__ Vt, const float* __restrict__ twt,
    __bf16* __restrict__ Ob) {
  __shared__ __bf16 sK[2][16][64];   // [buf][key][d]
  __shared__ __bf16 sV[2][64][16];   // [buf][d][key]
  __shared__ float  sP[4][16][17];

  const int tid  = threadIdx.x;
  const int wave = tid >> 5;
  const int lane = tid & 31;
  const int grp  = lane >> 4;
  const int lr   = lane & 15;
  const int sel  = grp << 3;

  const int blk   = blockIdx.x;             // b*NHEAD*16 + head*16 + qtile
  const int qtile = blk & 15;
  const int head  = (blk >> 4) % NHEAD;
  const int b     = blk / (16 * NHEAD);
  const int q0    = qtile * 64 + wave * 16;

  // Per-thread async-copy slots (block-wide cooperative staging)
  const int krow = tid >> 3, kdch = (tid & 7) * 8;   // K: 16 rows x 64 d
  const int vrow = tid >> 1, vkch = (tid & 1) * 8;   // V: 64 d   x 16 keys
  const __bf16* kgbase = Kb + ((size_t)(b * TKV + krow)) * DINO + head * HD + kdch;
  const __bf16* vgbase = Vt + ((size_t)(b * DINO + head * HD + vrow)) * TKV + vkch;

  // Temporal weights: one frame per 1024 keys; a 16-key chunk never straddles.
  const float w0 = 0.125f * twt[0];
  const float w1 = 0.125f * twt[1];
  const float w2 = 0.125f * twt[2];
  const float w3 = 0.125f * twt[3];

  // Q A-fragments for d = 0..31 and 32..63
  const __bf16* qrow = Q + ((size_t)(b * BHW + q0 + lr)) * DINO + head * HD;
  FragU qf0, qf1;
  qf0.h[0] = *(const bf16x8*)(qrow + sel);
  qf0.h[1] = *(const bf16x8*)(qrow + 16 + sel);
  qf1.h[0] = *(const bf16x8*)(qrow + 32 + sel);
  qf1.h[1] = *(const bf16x8*)(qrow + 48 + sel);

  f32x8 o[4] = {};
  float mi[8], li[8];
#pragma unroll
  for (int j = 0; j < 8; ++j) { mi[j] = -3.0e38f; li[j] = 0.f; }

  // Prologue: DMA chunk 0 into buffer 0.
  async_ld_b128(lds_off(&sK[0][krow][kdch]), kgbase);
  async_ld_b128(lds_off(&sV[0][vrow][vkch]), vgbase);

  for (int kt = 0; kt < TKV / 16; ++kt) {
    const int kb  = kt * 16;
    const int buf = kt & 1;

    // Current chunk's DMA complete + visible to all waves.
    wait_async0();
    __syncthreads();

    // K^T B-fragments from LDS: col = key (lane lr), rows = d contiguous.
    FragU kf0, kf1;
    kf0.h[0] = *(const bf16x8*)&sK[buf][lr][sel];
    kf0.h[1] = *(const bf16x8*)&sK[buf][lr][16 + sel];
    kf1.h[0] = *(const bf16x8*)&sK[buf][lr][32 + sel];
    kf1.h[1] = *(const bf16x8*)&sK[buf][lr][48 + sel];

    // Kick off next chunk's DMA into the other buffer (overlaps compute).
    if (kt + 1 < TKV / 16) {
      const int nb = kb + 16;
      async_ld_b128(lds_off(&sK[buf ^ 1][krow][kdch]), kgbase + (size_t)nb * DINO);
      async_ld_b128(lds_off(&sV[buf ^ 1][vrow][vkch]), vgbase + nb);
    }

    f32x8 s = {};
    s = wmma_bf16(qf0.v, kf0.v, s);
    s = wmma_bf16(qf1.v, kf1.v, s);

    const int frame = kb >> 10;              // uniform over the chunk
    const float wk = (frame == 0) ? w0 : (frame == 1) ? w1 : (frame == 2) ? w2 : w3;

    float p[8], sc[8];
#pragma unroll
    for (int j = 0; j < 8; ++j) {
      const float v = s[j] * wk;
      // row max across the 16 lanes holding this row (xor masks stay in-half)
      float r = v;
      r = fmaxf(r, __shfl_xor(r, 1));
      r = fmaxf(r, __shfl_xor(r, 2));
      r = fmaxf(r, __shfl_xor(r, 4));
      r = fmaxf(r, __shfl_xor(r, 8));
      const float mnew = fmaxf(mi[j], r);
      sc[j] = __expf(mi[j] - mnew);
      p[j]  = __expf(v - mnew);
      float rs = p[j];
      rs += __shfl_xor(rs, 1);
      rs += __shfl_xor(rs, 2);
      rs += __shfl_xor(rs, 4);
      rs += __shfl_xor(rs, 8);
      li[j] = li[j] * sc[j] + rs;
      mi[j] = mnew;
    }
#pragma unroll
    for (int nt = 0; nt < 4; ++nt)
#pragma unroll
      for (int j = 0; j < 8; ++j) o[nt][j] *= sc[j];

    // Transpose P: C layout (lane = col, vgpr = row) -> A layout (lane = row)
    __syncthreads();
#pragma unroll
    for (int j = 0; j < 8; ++j) sP[wave][(grp << 3) + j][lr] = p[j];
    __syncthreads();
    bf16x16 pf;
#pragma unroll
    for (int i = 0; i < 8; ++i) {
      pf[i]     = (__bf16)sP[wave][lr][sel + i];  // k = sel+i (valid 0..15)
      pf[i + 8] = (__bf16)0.f;                    // k = 16..31 zero-padded
    }

    // O += P(16x32, hi half zero) * V(32x16 per d-tile) from LDS.
    // The k=16..31 B half re-reads the same 16B (contributions are zeroed).
#pragma unroll
    for (int nt = 0; nt < 4; ++nt) {
      FragU vf;
      vf.h[0] = *(const bf16x8*)&sV[buf][nt * 16 + lr][sel];
      vf.h[1] = vf.h[0];
      o[nt] = wmma_bf16(pf, vf.v, o[nt]);
    }
  }

  // Normalize and store bf16 attention output rows [B*HW, 768]
#pragma unroll
  for (int nt = 0; nt < 4; ++nt) {
    const int c = head * HD + nt * 16 + lr;
#pragma unroll
    for (int j = 0; j < 8; ++j) {
      const int q = q0 + (grp << 3) + j;
      Ob[((size_t)(b * BHW + q)) * DINO + c] = (__bf16)(o[nt][j] / li[j]);
    }
  }
}

// ---------------------------------------------------------------------------
// Host launcher
// ---------------------------------------------------------------------------
extern "C" void kernel_launch(void* const* d_in, const int* in_sizes, int n_in,
                              void* d_out, int out_size, void* d_ws, size_t ws_size,
                              hipStream_t stream) {
  (void)in_sizes; (void)n_in; (void)out_size; (void)ws_size;

  const float* dino = (const float*)d_in[0];
  const float* cog  = (const float*)d_in[1];
  const float* twt  = (const float*)d_in[2];
  const float* Wq   = (const float*)d_in[3];
  const float* bq   = (const float*)d_in[4];
  const float* Wk   = (const float*)d_in[5];
  const float* bk   = (const float*)d_in[6];
  const float* Wv   = (const float*)d_in[7];
  const float* bv   = (const float*)d_in[8];
  const float* Wo   = (const float*)d_in[9];
  const float* bo   = (const float*)d_in[10];
  const float* gq   = (const float*)d_in[11];
  const float* Bq   = (const float*)d_in[12];
  const float* gkv  = (const float*)d_in[13];
  const float* Bkv  = (const float*)d_in[14];

  char* ws = (char*)d_ws;
  size_t off = 0;
  auto alloc = [&](size_t bytes) -> void* {
    void* p = ws + off;
    off = (off + bytes + 255) & ~(size_t)255;
    return p;
  };

  __bf16* xq  = (__bf16*)alloc((size_t)NB * BHW * DINO * 2);  // LN(dino)  [2048,768]
  __bf16* xkv = (__bf16*)alloc((size_t)NB * TKV * COG  * 2);  // LN(cog)   [8192,1920]
  __bf16* wqt = (__bf16*)alloc((size_t)DINO * DINO * 2);      // Wq^T [768,768]
  __bf16* wkt = (__bf16*)alloc((size_t)DINO * COG  * 2);      // Wk^T [768,1920]
  __bf16* wvt = (__bf16*)alloc((size_t)DINO * COG  * 2);      // Wv^T [768,1920]
  __bf16* wot = (__bf16*)alloc((size_t)DINO * DINO * 2);      // Wo^T [768,768]
  __bf16* Qb  = (__bf16*)alloc((size_t)NB * BHW * DINO * 2);  // [2048,768]
  __bf16* Kb  = (__bf16*)alloc((size_t)NB * TKV * DINO * 2);  // [8192,768]
  __bf16* Vt  = (__bf16*)alloc((size_t)NB * DINO * TKV * 2 + 256); // V^T + pad
  __bf16* Ab  = (__bf16*)alloc((size_t)NB * BHW * DINO * 2);  // attn out [2048,768]

  ln_dino_kernel<<<NB * BHW, 256, 0, stream>>>(dino, gq, Bq, xq);
  ln_cog_kernel <<<NB * TKV, 256, 0, stream>>>(cog, gkv, Bkv, xkv);

  // Transposed bf16 weights (W[K,N] -> WT[N,K])
  cvtT_kernel<<<dim3(DINO / 32, DINO / 32), dim3(32, 8), 0, stream>>>(Wq, wqt, DINO, DINO);
  cvtT_kernel<<<dim3(COG  / 32, DINO / 32), dim3(32, 8), 0, stream>>>(Wk, wkt, COG,  DINO);
  cvtT_kernel<<<dim3(COG  / 32, DINO / 32), dim3(32, 8), 0, stream>>>(Wv, wvt, COG,  DINO);
  cvtT_kernel<<<dim3(DINO / 32, DINO / 32), dim3(32, 8), 0, stream>>>(Wo, wot, DINO, DINO);

  gemm_kernel<0><<<dim3((NB * BHW) / 128, DINO / 64), 128, 0, stream>>>(
      xq, wqt, bq, (void*)Qb, NB * BHW, DINO, DINO);
  gemm_kernel<0><<<dim3((NB * TKV) / 128, DINO / 64), 128, 0, stream>>>(
      xkv, wkt, bk, (void*)Kb, NB * TKV, DINO, COG);
  gemm_kernel<1><<<dim3((NB * TKV) / 128, DINO / 64), 128, 0, stream>>>(
      xkv, wvt, bv, (void*)Vt, NB * TKV, DINO, COG);

  attn_kernel<<<NB * NHEAD * (BHW / 64), 128, 0, stream>>>(Qb, Kb, Vt, twt, Ab);

  gemm_kernel<2><<<dim3((NB * BHW) / 128, DINO / 64), 128, 0, stream>>>(
      Ab, wot, bo, d_out, NB * BHW, DINO, DINO);
}
